// MultimodalRegionAwareAttention_2757369004162
// MI455X (gfx1250) — compile-verified, matrix operand order checked by reading, not run on hardware
//
#include <hip/hip_runtime.h>
#include <math.h>

// MI455X / gfx1250: wave32, WMMA f32_16x16x32_f16, flash-style fused
// multimodal region-gathered attention. One 128-thread block (4 waves)
// per (q_modality, batch, head, query_region) tile; 2048 blocks total.
// Round 2: DPP16 butterfly reductions (no ds_bpermute on the LDS pipe),
// global_prefetch_b8 of the next K/V chunk overlapped with compute.

typedef _Float16 v16h __attribute__((ext_vector_type(16)));
typedef float    v8f  __attribute__((ext_vector_type(8)));

#define PITCH 72          // halves per LDS row: 144B = 36 dwords -> conflict-friendly, 16B aligned
#define QK_SCALE 0.125f   // HEAD_DIM^-0.5 = 64^-0.5

__device__ __forceinline__ int pos_off(int pos) {
    // pos in [0,64) -> spatial offset within a 4x4x4 region of a 16^3 grid
    return ((pos >> 4) << 8) | (((pos >> 2) & 3) << 4) | (pos & 3);
}

// ---- DPP16 cross-lane helpers (reduce within each 16-lane half-wave) ----
template <int CTRL>
__device__ __forceinline__ float dpp16_f(float x) {
    return __builtin_bit_cast(float,
        __builtin_amdgcn_update_dpp(0, __builtin_bit_cast(int, x),
                                    CTRL, 0xF, 0xF, true));
}
__device__ __forceinline__ float red_max16(float x) {
    x = fmaxf(x, dpp16_f<0xB1>(x));    // quad_perm(1,0,3,2)  : xor 1
    x = fmaxf(x, dpp16_f<0x4E>(x));    // quad_perm(2,3,0,1)  : xor 2
    x = fmaxf(x, dpp16_f<0x141>(x));   // row_half_mirror     : folds 8-groups
    x = fmaxf(x, dpp16_f<0x140>(x));   // row_mirror          : folds 16-row
    return x;
}
__device__ __forceinline__ float red_sum16(float x) {
    x += dpp16_f<0xB1>(x);
    x += dpp16_f<0x4E>(x);
    x += dpp16_f<0x141>(x);
    x += dpp16_f<0x140>(x);
    return x;
}

__global__ __launch_bounds__(128)
void mmra_attn_kernel(
    const int*   __restrict__ mask,
    const float* __restrict__ q0, const float* __restrict__ q1,
    const float* __restrict__ q2, const float* __restrict__ q3,
    const float* __restrict__ k0, const float* __restrict__ k1,
    const float* __restrict__ k2, const float* __restrict__ k3,
    const float* __restrict__ v0, const float* __restrict__ v1,
    const float* __restrict__ v2, const float* __restrict__ v3,
    const int*   __restrict__ rg0, const int*  __restrict__ rg1,
    const int*   __restrict__ rg2, const int*  __restrict__ rg3,
    float* __restrict__ out)
{
    __shared__ _Float16 Qs[64 * PITCH];   // 9216 B each
    __shared__ _Float16 Ks[64 * PITCH];
    __shared__ _Float16 Vs[64 * PITCH];
    __shared__ _Float16 Ps[64 * PITCH];

    const float* const qt[4] = {q0, q1, q2, q3};
    const float* const kt[4] = {k0, k1, k2, k3};
    const float* const vt[4] = {v0, v1, v2, v3};
    const int*   const rt[4] = {rg0, rg1, rg2, rg3};

    const int tile = blockIdx.x;          // ((mq*2 + b)*4 + h)*64 + r
    const int r    = tile & 63;
    const int h    = (tile >> 6) & 3;
    const int b    = (tile >> 8) & 1;
    const int mq   = (tile >> 9) & 3;

    const int tid  = threadIdx.x;
    const int w    = tid >> 5;            // wave id 0..3 -> 16-row query band
    const int lane = tid & 31;
    const int l16  = lane & 15;
    const int hi   = lane >> 4;           // half-wave select

    const int  qspb = ((r >> 4) << 10) | (((r >> 2) & 3) << 6) | ((r & 3) << 2);
    const long bh   = (long)(b * 256 + h * 64) * 4096;
    const int  rgbase = (((b * 4 + h) << 6) + r) * 4;

    // ---- load Q band tile (f32 -> f16, fold in 1/sqrt(d)) ----
    const float* qp = qt[mq];
    for (int i = tid; i < 4096; i += 128) {
        int c = i >> 6, pos = i & 63;
        Qs[pos * PITCH + c] = (_Float16)(qp[bh + qspb + (long)c * 4096 + pos_off(pos)] * QK_SCALE);
    }
    __syncthreads();

    // ---- Q A-fragments (16x32 f16 A-layout: row=l16, K=(i/8)*16 + hi*8 + i%8) ----
    v16h qA[2];
    #pragma unroll
    for (int ks = 0; ks < 2; ++ks)
        #pragma unroll
        for (int i = 0; i < 16; ++i)
            qA[ks][i] = Qs[(w * 16 + l16) * PITCH + ks * 32 + (i >> 3) * 16 + hi * 8 + (i & 7)];

    float mst[8], lst[8];
    v8f   o[4];
    #pragma unroll
    for (int v = 0; v < 8; ++v) { mst[v] = -INFINITY; lst[v] = 0.f; }
    #pragma unroll
    for (int ot = 0; ot < 4; ++ot)
        #pragma unroll
        for (int v = 0; v < 8; ++v) o[ot][v] = 0.f;

    // ---- stream 16 gathered K/V chunks (4 kv-modalities x top-4 regions) ----
    for (int ci = 0; ci < 16; ++ci) {
        const int mod = ci >> 2, tk = ci & 3;
        if (mask[b * 4 + mod] == 0) continue;                       // block-uniform skip
        const int kr  = rt[mod][rgbase + tk];
        const int spb = ((kr >> 4) << 10) | (((kr >> 2) & 3) << 6) | ((kr & 3) << 2);
        const float* kp = kt[mod];
        const float* vp = vt[mod];

        __syncthreads();
        for (int i = tid; i < 4096; i += 128) {
            int c = i >> 6, pos = i & 63;
            long g = bh + spb + (long)c * 4096 + pos_off(pos);
            Ks[pos * PITCH + c] = (_Float16)kp[g];
            Vs[pos * PITCH + c] = (_Float16)vp[g];
        }
        __syncthreads();

        // Prefetch next chunk while this one computes.
        // idx = i*8 => c = i>>3, pos = 8*(i&7): hits exactly the 8 distinct
        // 128B cachelines per channel that the region gather touches.
        if (ci + 1 < 16) {
            const int nmod = (ci + 1) >> 2, ntk = (ci + 1) & 3;
            const int nkr  = rt[nmod][rgbase + ntk];
            const int nspb = ((nkr >> 4) << 10) | (((nkr >> 2) & 3) << 6) | ((nkr & 3) << 2);
            const float* nkp = kt[nmod];
            const float* nvp = vt[nmod];
            #pragma unroll
            for (int e = 0; e < 4; ++e) {
                int i   = tid + (e << 7);            // 0..511
                int c   = i >> 3;
                int pos = (i & 7) << 3;
                long g  = bh + nspb + (long)c * 4096 + pos_off(pos);
                __builtin_prefetch(&nkp[g], 0, 3);
                __builtin_prefetch(&nvp[g], 0, 3);
            }
        }

        // S = Q * K^T for this wave's 16x64 band (B-frag of K^T == A-style read of K)
        v8f s[4];
        #pragma unroll
        for (int t = 0; t < 4; ++t) {
            v8f acc;
            #pragma unroll
            for (int v = 0; v < 8; ++v) acc[v] = 0.f;
            #pragma unroll
            for (int ks = 0; ks < 2; ++ks) {
                v16h bf;
                #pragma unroll
                for (int i = 0; i < 16; ++i)
                    bf[i] = Ks[(t * 16 + l16) * PITCH + ks * 32 + (i >> 3) * 16 + hi * 8 + (i & 7)];
                acc = __builtin_amdgcn_wmma_f32_16x16x32_f16(false, qA[ks], false, bf,
                                                             (short)0, acc, false, false);
            }
            s[t] = acc;
        }

        // online softmax: rows stripe as v + 8*hi; columns across the 16-lane half-group
        float scl[8];
        #pragma unroll
        for (int v = 0; v < 8; ++v) {
            float cm = fmaxf(fmaxf(s[0][v], s[1][v]), fmaxf(s[2][v], s[3][v]));
            cm = red_max16(cm);
            float mnew = fmaxf(mst[v], cm);
            float sc   = __expf(mst[v] - mnew);                      // exp(-inf)=0 first time
            float rs   = 0.f;
            #pragma unroll
            for (int t = 0; t < 4; ++t) {
                float p = __expf(s[t][v] - mnew);
                s[t][v] = p;
                rs += p;
            }
            rs = red_sum16(rs);
            lst[v] = lst[v] * sc + rs;
            mst[v] = mnew;
            scl[v] = sc;
        }
        #pragma unroll
        for (int ot = 0; ot < 4; ++ot)
            #pragma unroll
            for (int v = 0; v < 8; ++v) o[ot][v] *= scl[v];

        // round-trip P through LDS: C-fragment layout out, A-fragment layout back in
        #pragma unroll
        for (int t = 0; t < 4; ++t)
            #pragma unroll
            for (int v = 0; v < 8; ++v)
                Ps[(w * 16 + v + 8 * hi) * PITCH + t * 16 + l16] = (_Float16)s[t][v];
        __syncthreads();

        v16h pA[2];
        #pragma unroll
        for (int ks = 0; ks < 2; ++ks)
            #pragma unroll
            for (int i = 0; i < 16; ++i)
                pA[ks][i] = Ps[(w * 16 + l16) * PITCH + ks * 32 + (i >> 3) * 16 + hi * 8 + (i & 7)];

        // O += P * V  (B-frag of V: col = l16, K striped per half-wave)
        #pragma unroll
        for (int ot = 0; ot < 4; ++ot)
            #pragma unroll
            for (int ks = 0; ks < 2; ++ks) {
                v16h vB;
                #pragma unroll
                for (int i = 0; i < 16; ++i) {
                    int kk = ks * 32 + (i >> 3) * 16 + hi * 8 + (i & 7);
                    vB[i] = Vs[kk * PITCH + ot * 16 + l16];
                }
                o[ot] = __builtin_amdgcn_wmma_f32_16x16x32_f16(false, pA[ks], false, vB,
                                                               (short)0, o[ot], false, false);
            }
    }

    // ---- normalize and scatter back to [M,B,C,16,16,16] grid layout ----
    float inv[8];
    #pragma unroll
    for (int v = 0; v < 8; ++v) inv[v] = (lst[v] > 0.f) ? (1.0f / lst[v]) : 0.f;

    const long ob = (long)(mq * 2 + b) * 1048576 + (long)(h * 64) * 4096 + qspb;
    #pragma unroll
    for (int ot = 0; ot < 4; ++ot)
        #pragma unroll
        for (int v = 0; v < 8; ++v) {
            int pos = w * 16 + v + 8 * hi;
            int c   = ot * 16 + l16;
            out[ob + (long)c * 4096 + pos_off(pos)] = o[ot][v] * inv[v];
        }
}

extern "C" void kernel_launch(void* const* d_in, const int* in_sizes, int n_in,
                              void* d_out, int out_size, void* d_ws, size_t ws_size,
                              hipStream_t stream) {
    (void)in_sizes; (void)n_in; (void)out_size; (void)d_ws; (void)ws_size;
    const int* mask = (const int*)d_in[0];
    const float *q[4], *k[4], *v[4];
    const int *rg[4];
    for (int i = 0; i < 4; ++i) {
        q[i]  = (const float*)d_in[1 + i];
        k[i]  = (const float*)d_in[5 + i];
        v[i]  = (const float*)d_in[9 + i];
        rg[i] = (const int*)d_in[13 + i];
    }
    dim3 grid(2048), block(128);
    mmra_attn_kernel<<<grid, block, 0, stream>>>(
        mask,
        q[0], q[1], q[2], q[3],
        k[0], k[1], k[2], k[3],
        v[0], v[1], v[2], v[3],
        rg[0], rg[1], rg[2], rg[3],
        (float*)d_out);
}
